// Multi_Head_Attention_For_Dynamic_26061861552843
// MI455X (gfx1250) — matrix operand order repaired and checked by reading, not compiled, not run on hardware
//
#include <hip/hip_runtime.h>
#include <hip/hip_bf16.h>

// ---------------------------------------------------------------------------
// MI455X (gfx1250, wave32) implementation.
// - All matmuls: v_wmma_f32_16x16x32_f16 (f16 in, f32 accumulate).
// - GEMM weight panel staged via async Global->LDS (double-buffered,
//   global_load_async_to_lds_b128 + s_wait_asynccnt), shared by 8 waves.
// - Attention is flash-style (online softmax); logits never hit memory.
// ---------------------------------------------------------------------------

typedef _Float16 half_t;
typedef __attribute__((ext_vector_type(16))) _Float16 v16h;
typedef __attribute__((ext_vector_type(8)))  float    v8f;

union V16H { v16h v; uint4 u[2]; };

#define WMMA_F16(a, b, c) \
  __builtin_amdgcn_wmma_f32_16x16x32_f16(false, (a), false, (b), (short)0, (c), false, false)

static constexpr int   cB  = 4;
static constexpr int   cN  = 2048;
static constexpr int   cM  = 1024;
static constexpr int   cD  = 512;
static constexpr int   cH  = 8;
static constexpr int   cDK = 64;
static constexpr float SCALEF = 0.125f;          // 1/sqrt(64)
static constexpr float NEGV   = -9.0e15f;

__device__ __forceinline__ float gelu_f(float x) {
  return 0.5f * x * (1.0f + erff(x * 0.70710678118654752f));
}

__device__ __forceinline__ v8f vzero8() {
  v8f v = {0.f, 0.f, 0.f, 0.f, 0.f, 0.f, 0.f, 0.f};
  return v;
}

// ---------------------------------------------------------------------------
// f32 -> f16 convert
// ---------------------------------------------------------------------------
__global__ void cvt_f16_kernel(const float* __restrict__ in, half_t* __restrict__ out,
                               size_t n) {
  size_t i = (size_t)blockIdx.x * blockDim.x + threadIdx.x;
  if (i < n) out[i] = (half_t)in[i];
}

// 512x512 weight: W[k][n] -> WT[n][k] (f16), so GEMM B-frags read K-contiguous rows.
__global__ void wtrans_kernel(const float* __restrict__ W, half_t* __restrict__ WT) {
  __shared__ float t[16][17];
  int tx = threadIdx.x, ty = threadIdx.y;
  int k = blockIdx.y * 16 + ty;
  int n = blockIdx.x * 16 + tx;
  t[ty][tx] = W[(size_t)k * cD + n];
  __syncthreads();
  int nn = blockIdx.x * 16 + ty;
  int kk = blockIdx.y * 16 + tx;
  WT[(size_t)nn * cD + kk] = (half_t)t[tx][ty];
}

// ---------------------------------------------------------------------------
// GEMM: C[rows,512] = A[rows,512] @ W[512,512] + bias, optional exact GELU.
// A f16 row-major, BT = W^T f16 row-major [512out][512in].
// Block = 256 thr = 8 waves; wave -> 16x64 tile; block tile 128x64.
// The 64x32 B chunk per K-step is shared by all waves: staged into LDS with
// double-buffered GLOBAL_LOAD_ASYNC_TO_LDS_B128 (ASYNCcnt), transfer of
// chunk i+1 overlapped with WMMAs on chunk i.
// Optional outputs: f32 C, f16 C, f16 transposed [batch, 512, rowsPerBatch].
// ---------------------------------------------------------------------------
__global__ __launch_bounds__(256) void gemm512_kernel(
    const half_t* __restrict__ A, const half_t* __restrict__ BT,
    const float* __restrict__ bias,
    float* __restrict__ Cf, half_t* __restrict__ Ch, half_t* __restrict__ ChT,
    int rowsPerBatch, int applyGelu) {
  __shared__ __align__(16) half_t sB[2][64 * 32];   // 2 x 4KB B chunks

  const int tid   = threadIdx.x;
  const int wave  = tid >> 5;
  const int lane  = tid & 31;
  const int lmod  = lane & 15;
  const int lhalf = lane >> 4;
  const int m0 = blockIdx.x * 128 + wave * 16;
  const int n0 = blockIdx.y * 64;

  // Async staging: thread t moves 16B: row = t>>2 (of 64), quarter = t&3.
  const int srow = tid >> 2;
  const int sq   = tid & 3;
  const half_t* gsrc = BT + (size_t)(n0 + srow) * cD + sq * 8;  // + kk per chunk
  const unsigned int ldsOff0 =
      (unsigned int)(uintptr_t)(&sB[0][srow * 32 + sq * 8]);
  const unsigned int ldsOff1 =
      (unsigned int)(uintptr_t)(&sB[1][srow * 32 + sq * 8]);

  // preload chunk 0 into buffer 0
  {
    unsigned long long ga = (unsigned long long)(uintptr_t)gsrc;
    asm volatile("global_load_async_to_lds_b128 %0, %1, off"
                 :: "v"(ldsOff0), "v"(ga) : "memory");
  }

  v8f acc[4];
#pragma unroll
  for (int dt = 0; dt < 4; ++dt) acc[dt] = vzero8();

  const half_t* arow = A + ((size_t)m0 + lmod) * cD;

  for (int i = 0; i < 16; ++i) {
    // chunk i has landed in LDS (per-wave wait, then block-wide barrier; the
    // barrier also drains DS reads so the buffer overwritten below is free).
    asm volatile("s_wait_asynccnt 0x0" ::: "memory");
    __syncthreads();
    if (i < 15) {
      unsigned long long ga =
          (unsigned long long)(uintptr_t)(gsrc + (i + 1) * 32);
      unsigned int dst = ((i + 1) & 1) ? ldsOff1 : ldsOff0;
      asm volatile("global_load_async_to_lds_b128 %0, %1, off"
                   :: "v"(dst), "v"(ga) : "memory");
    }

    const int kk = i * 32;
    V16H a;
    const half_t* ap = arow + kk + lhalf * 8;
    a.u[0] = *(const uint4*)ap;
    a.u[1] = *(const uint4*)(ap + 16);

    const half_t* bbase = &sB[i & 1][0];
#pragma unroll
    for (int dt = 0; dt < 4; ++dt) {
      const half_t* bp = bbase + (dt * 16 + lmod) * 32 + lhalf * 8;
      V16H bb;
      bb.u[0] = *(const uint4*)bp;
      bb.u[1] = *(const uint4*)(bp + 16);
      acc[dt] = WMMA_F16(a.v, bb.v, acc[dt]);
    }
  }

#pragma unroll
  for (int dt = 0; dt < 4; ++dt) {
#pragma unroll
    for (int r = 0; r < 8; ++r) {
      int m = m0 + lhalf * 8 + r;
      int n = n0 + dt * 16 + lmod;
      float v = acc[dt][r] + bias[n];
      if (applyGelu) v = gelu_f(v);
      if (Cf) Cf[(size_t)m * cD + n] = v;
      if (Ch) Ch[(size_t)m * cD + n] = (half_t)v;
      if (ChT) {
        int bb_ = m / rowsPerBatch;
        int lr_ = m - bb_ * rowsPerBatch;
        ChT[((size_t)bb_ * cD + n) * rowsPerBatch + lr_] = (half_t)v;
      }
    }
  }
}

// ---------------------------------------------------------------------------
// Flash attention: Out[b,q,h*64+d] = gelu( softmax_k( mask(Q.K^T*s) ) @ V )
//   Q  : [B, LQ, 512] f16      (head h = cols h*64..h*64+63)
//   Km : [B, LK, 512] f16
//   VT : [B, 512, LK] f16      (transposed V, K-contiguous for B-frags)
//   inc: [B, N, M] int32.  MASK_QK ? inc[b][q][k] : inc[b][k][q]
// One wave per 16-row query tile; softmax axis walked in 32-wide chunks.
// ---------------------------------------------------------------------------
template <int LQ, int LK, bool MASK_QK>
__global__ __launch_bounds__(256) void attn_kernel(
    const half_t* __restrict__ Q, const half_t* __restrict__ Km,
    const half_t* __restrict__ VT, const int* __restrict__ inc,
    float* __restrict__ Out) {
  __shared__ __align__(16) half_t sP[8][16 * 32];

  const int wave  = threadIdx.x >> 5;
  const int lane  = threadIdx.x & 31;
  const int lmod  = lane & 15;
  const int lhalf = lane >> 4;
  const int b = blockIdx.z;
  const int h = blockIdx.y;
  const int q0 = (blockIdx.x * 8 + wave) * 16;

  // Q fragments for DK=64 (two K=32 chunks), loaded once.
  const half_t* qrow = Q + ((size_t)b * LQ + q0 + lmod) * cD + h * cDK;
  V16H qa0, qa1;
  qa0.u[0] = *(const uint4*)(qrow + lhalf * 8);
  qa0.u[1] = *(const uint4*)(qrow + lhalf * 8 + 16);
  qa1.u[0] = *(const uint4*)(qrow + 32 + lhalf * 8);
  qa1.u[1] = *(const uint4*)(qrow + 32 + lhalf * 8 + 16);

  const half_t* kbase = Km + (size_t)b * LK * cD + h * cDK;
  const half_t* vbase = VT + ((size_t)b * cD + h * cDK) * LK;
  const int* incb = inc + (size_t)b * LQ * LK;  // N*M either way

  float mr[8], lr[8];
  v8f o[4];
#pragma unroll
  for (int r = 0; r < 8; ++r) { mr[r] = -3.0e38f; lr[r] = 0.f; }
#pragma unroll
  for (int dt = 0; dt < 4; ++dt) o[dt] = vzero8();

  for (int k0 = 0; k0 < LK; k0 += 32) {
    // ---- logits: two 16x16 S tiles (K dim = DK=64 -> 2 WMMAs each) ----
    v8f s[2];
#pragma unroll
    for (int t = 0; t < 2; ++t) {
      const half_t* kp = kbase + (size_t)(k0 + t * 16 + lmod) * cD + lhalf * 8;
      V16H b0, b1;
      b0.u[0] = *(const uint4*)kp;
      b0.u[1] = *(const uint4*)(kp + 16);
      b1.u[0] = *(const uint4*)(kp + 32);
      b1.u[1] = *(const uint4*)(kp + 48);
      v8f st = vzero8();
      st = WMMA_F16(qa0.v, b0.v, st);
      st = WMMA_F16(qa1.v, b1.v, st);
      s[t] = st;
    }
    // ---- scale + incidence mask ----
#pragma unroll
    for (int t = 0; t < 2; ++t) {
      int kidx = k0 + t * 16 + lmod;
#pragma unroll
      for (int r = 0; r < 8; ++r) {
        int qidx = q0 + lhalf * 8 + r;
        size_t off = MASK_QK ? ((size_t)qidx * LK + kidx)
                             : ((size_t)kidx * LQ + qidx);
        float sv = s[t][r] * SCALEF;
        s[t][r] = (incb[off] > 0) ? sv : NEGV;
      }
    }
    // ---- online softmax: per-row max/sum via 16-lane xor reductions ----
    float al[8];
#pragma unroll
    for (int r = 0; r < 8; ++r) {
      float bmv = fmaxf(s[0][r], s[1][r]);
      bmv = fmaxf(bmv, __shfl_xor(bmv, 8));
      bmv = fmaxf(bmv, __shfl_xor(bmv, 4));
      bmv = fmaxf(bmv, __shfl_xor(bmv, 2));
      bmv = fmaxf(bmv, __shfl_xor(bmv, 1));
      float mn = fmaxf(mr[r], bmv);
      al[r] = __expf(mr[r] - mn);
      mr[r] = mn;
      float p0 = __expf(s[0][r] - mn);
      float p1 = __expf(s[1][r] - mn);
      sP[wave][(lhalf * 8 + r) * 32 + lmod]      = (half_t)p0;
      sP[wave][(lhalf * 8 + r) * 32 + 16 + lmod] = (half_t)p1;
      float pr = p0 + p1;
      pr += __shfl_xor(pr, 8);
      pr += __shfl_xor(pr, 4);
      pr += __shfl_xor(pr, 2);
      pr += __shfl_xor(pr, 1);
      lr[r] = lr[r] * al[r] + pr;
    }
#pragma unroll
    for (int dt = 0; dt < 4; ++dt)
#pragma unroll
      for (int r = 0; r < 8; ++r) o[dt][r] *= al[r];

    // ---- reshape P (D-frag f32 -> A-frag f16) through wave-private LDS ----
    const half_t* pp = &sP[wave][lmod * 32 + lhalf * 8];
    V16H pa;
    pa.u[0] = *(const uint4*)pp;
    pa.u[1] = *(const uint4*)(pp + 16);

    // ---- accumulate P @ V into 16x64 output (4 d-tiles) ----
#pragma unroll
    for (int dt = 0; dt < 4; ++dt) {
      const half_t* vp = vbase + (size_t)(dt * 16 + lmod) * LK + k0 + lhalf * 8;
      V16H vb;
      vb.u[0] = *(const uint4*)vp;
      vb.u[1] = *(const uint4*)(vp + 16);
      o[dt] = WMMA_F16(pa.v, vb.v, o[dt]);
    }
  }

  // ---- epilogue: normalize, exact GELU, merged-head store ----
#pragma unroll
  for (int dt = 0; dt < 4; ++dt) {
#pragma unroll
    for (int r = 0; r < 8; ++r) {
      int qi = q0 + lhalf * 8 + r;
      int d  = h * cDK + dt * 16 + lmod;
      float v = o[dt][r] / lr[r];
      Out[((size_t)b * LQ + qi) * cD + d] = gelu_f(v);
    }
  }
}

// ---------------------------------------------------------------------------
// LayerNorm(E + Ae) -> f16.  One wave per row (512 cols = 16/lane).
// ---------------------------------------------------------------------------
__global__ __launch_bounds__(256) void ln_kernel(
    const float* __restrict__ E0, const float* __restrict__ Ae,
    const float* __restrict__ g, const float* __restrict__ bta,
    half_t* __restrict__ outh) {
  const int wave = threadIdx.x >> 5;
  const int lane = threadIdx.x & 31;
  const int row  = blockIdx.x * 8 + wave;
  const float* p0 = E0 + (size_t)row * cD + lane * 16;
  const float* p1 = Ae + (size_t)row * cD + lane * 16;
  float x[16];
  float sum = 0.f;
#pragma unroll
  for (int i = 0; i < 16; ++i) { x[i] = p0[i] + p1[i]; sum += x[i]; }
  sum += __shfl_xor(sum, 16);
  sum += __shfl_xor(sum, 8);
  sum += __shfl_xor(sum, 4);
  sum += __shfl_xor(sum, 2);
  sum += __shfl_xor(sum, 1);
  float mu = sum * (1.0f / 512.0f);
  float ss = 0.f;
#pragma unroll
  for (int i = 0; i < 16; ++i) { float d = x[i] - mu; ss += d * d; }
  ss += __shfl_xor(ss, 16);
  ss += __shfl_xor(ss, 8);
  ss += __shfl_xor(ss, 4);
  ss += __shfl_xor(ss, 2);
  ss += __shfl_xor(ss, 1);
  float inv = rsqrtf(ss * (1.0f / 512.0f) + 1e-7f);
#pragma unroll
  for (int i = 0; i < 16; ++i) {
    int c = lane * 16 + i;
    outh[(size_t)row * cD + c] = (half_t)((x[i] - mu) * inv * g[c] + bta[c]);
  }
}

// ---------------------------------------------------------------------------
// Host orchestration
// ---------------------------------------------------------------------------
extern "C" void kernel_launch(void* const* d_in, const int* in_sizes, int n_in,
                              void* d_out, int out_size, void* d_ws, size_t ws_size,
                              hipStream_t stream) {
  (void)in_sizes; (void)n_in; (void)out_size; (void)ws_size;

  const float* X   = (const float*)d_in[0];
  const int*   inc = (const int*)  d_in[1];
  const float* E   = (const float*)d_in[3];
  const float* Wqn = (const float*)d_in[4];  const float* bqn = (const float*)d_in[5];
  const float* Wkn = (const float*)d_in[6];  const float* bkn = (const float*)d_in[7];
  const float* Wvn = (const float*)d_in[8];  const float* bvn = (const float*)d_in[9];
  const float* Wqe = (const float*)d_in[10]; const float* bqe = (const float*)d_in[11];
  const float* Wke = (const float*)d_in[12]; const float* bke = (const float*)d_in[13];
  const float* Wve = (const float*)d_in[14]; const float* bve = (const float*)d_in[15];
  const float* Wm1 = (const float*)d_in[16]; const float* bm1 = (const float*)d_in[17];
  const float* Wm2 = (const float*)d_in[18]; const float* bm2 = (const float*)d_in[19];
  const float* g_ln = (const float*)d_in[20];
  const float* b_ln = (const float*)d_in[21];

  const size_t BND = (size_t)cB * cN * cD;   // 4,194,304
  const size_t BMD = (size_t)cB * cM * cD;   // 2,097,152
  const size_t DD  = (size_t)cD * cD;        //   262,144

  char* cur = (char*)d_ws;
  auto allocH = [&](size_t n) { half_t* p = (half_t*)cur; cur += n * sizeof(half_t); return p; };
  auto allocF = [&](size_t n) { float*  p = (float*) cur; cur += n * sizeof(float);  return p; };

  half_t* Xh   = allocH(BND);
  half_t* Eh   = allocH(BMD);
  half_t* WTqn = allocH(DD); half_t* WTkn = allocH(DD); half_t* WTvn = allocH(DD);
  half_t* WTqe = allocH(DD); half_t* WTke = allocH(DD); half_t* WTve = allocH(DD);
  half_t* WTm1 = allocH(DD); half_t* WTm2 = allocH(DD);
  half_t* qn   = allocH(BND);
  half_t* kn   = allocH(BND);
  half_t* vnT  = allocH(BND);
  half_t* qe   = allocH(BMD);
  half_t* Eln  = allocH(BMD);
  half_t* H1   = allocH(BMD);
  half_t* Eh2  = allocH(BMD);
  half_t* ke   = allocH(BMD);
  half_t* veT  = allocH(BMD);
  float*  Ae   = allocF(BMD);

  float* outX = (float*)d_out;        // X_ : [4,2048,512]
  float* outE = outX + BND;           // E_ : [4,1024,512]

  // 1) converts + weight transposes
  cvt_f16_kernel<<<(int)(BND / 256), 256, 0, stream>>>(X, Xh, BND);
  cvt_f16_kernel<<<(int)(BMD / 256), 256, 0, stream>>>(E, Eh, BMD);
  {
    dim3 g(32, 32), t(16, 16);
    wtrans_kernel<<<g, t, 0, stream>>>(Wqn, WTqn);
    wtrans_kernel<<<g, t, 0, stream>>>(Wkn, WTkn);
    wtrans_kernel<<<g, t, 0, stream>>>(Wvn, WTvn);
    wtrans_kernel<<<g, t, 0, stream>>>(Wqe, WTqe);
    wtrans_kernel<<<g, t, 0, stream>>>(Wke, WTke);
    wtrans_kernel<<<g, t, 0, stream>>>(Wve, WTve);
    wtrans_kernel<<<g, t, 0, stream>>>(Wm1, WTm1);
    wtrans_kernel<<<g, t, 0, stream>>>(Wm2, WTm2);
  }

  // 2) projections (rows = B*N = 8192 -> 64 blocks; B*M = 4096 -> 32 blocks)
  dim3 gN(64, 8), gM(32, 8);
  gemm512_kernel<<<gN, 256, 0, stream>>>(Xh, WTqn, bqn, nullptr, qn, nullptr, cN, 0);
  gemm512_kernel<<<gN, 256, 0, stream>>>(Xh, WTkn, bkn, nullptr, kn, nullptr, cN, 0);
  gemm512_kernel<<<gN, 256, 0, stream>>>(Xh, WTvn, bvn, nullptr, nullptr, vnT, cN, 0);
  gemm512_kernel<<<gM, 256, 0, stream>>>(Eh, WTqe, bqe, nullptr, qe, nullptr, cM, 0);

  // 3) edge attends nodes: softmax over N, mask = inc[b][k][q]; fused GELU
  attn_kernel<cM, cN, false><<<dim3(8, 8, 4), 256, 0, stream>>>(qe, kn, vnT, inc, Ae);

  // 4) LayerNorm(E + Ae) -> f16   (B*M = 4096 rows, 8 rows/block)
  ln_kernel<<<512, 256, 0, stream>>>(E, Ae, g_ln, b_ln, Eln);

  // 5) MLP: H1 = gelu(Eln@Wm1+b1);  E_ = H1@Wm2+b2 -> d_out (f32) + f16 copy
  gemm512_kernel<<<gM, 256, 0, stream>>>(Eln, WTm1, bm1, nullptr, H1, nullptr, cM, 1);
  gemm512_kernel<<<gM, 256, 0, stream>>>(H1, WTm2, bm2, outE, Eh2, nullptr, cM, 0);

  // 6) edge K/V projections from final E_
  gemm512_kernel<<<gM, 256, 0, stream>>>(Eh2, WTke, bke, nullptr, ke, nullptr, cM, 0);
  gemm512_kernel<<<gM, 256, 0, stream>>>(Eh2, WTve, bve, nullptr, nullptr, veT, cM, 0);

  // 7) node attends edges: softmax over M, mask = inc[b][q][k]; fused GELU -> X_
  attn_kernel<cN, cM, true><<<dim3(16, 8, 4), 256, 0, stream>>>(qn, ke, veT, inc, outX);
}
